// Block_38053410242668
// MI455X (gfx1250) — matrix operand order repaired and checked by reading, not compiled, not run on hardware
//
#include <hip/hip_runtime.h>
#include <math.h>
#include <stdint.h>

// ---------------- problem constants ----------------
#define C     768
#define NH    12
#define BATCH 4
#define TT    16
#define KTOK  196
#define HD    64
#define LSP   197

constexpr int M_T   = BATCH * KTOK * TT;        // 12544 temporal tokens
constexpr int NSEQT = BATCH * KTOK;             // 784 temporal sequences
constexpr int NSEQS = BATCH * TT;               // 64 spatial sequences
constexpr int M_S   = NSEQS * LSP;              // 12608 spatial tokens
constexpr int M_F   = BATCH * (KTOK * TT + 1);  // 12548 full tokens

// ---------------- WMMA plumbing (wave32, gfx1250) ----------------
typedef __attribute__((ext_vector_type(16))) __bf16 v16bf;
typedef __attribute__((ext_vector_type(8)))  float  v8f;

union ABfrag { v16bf v; uint4 u4[2]; unsigned short s[16]; };

__device__ __forceinline__ unsigned short f2bf(float f) {
  __bf16 h = (__bf16)f;                    // native v_cvt path
  return __builtin_bit_cast(unsigned short, h);
}

__device__ __forceinline__ v8f wmma_bf16(ABfrag a, ABfrag b, v8f c) {
  return __builtin_amdgcn_wmma_f32_16x16x32_bf16(
      false, a.v, false, b.v, (short)0, c, false, false);
}

__device__ __forceinline__ float gelu_exact(float x) {
  return 0.5f * x * (1.0f + erff(x * 0.70710678118654752f));
}

// CDNA5 async global->LDS DMA (ASYNCcnt tracked), 16B per lane.
__device__ __forceinline__ void async_b128(unsigned lds_byte_off, const unsigned short* g) {
  asm volatile("global_load_async_to_lds_b128 %0, %1, off"
               :: "v"(lds_byte_off), "v"(g) : "memory");
}
__device__ __forceinline__ void wait_async0() {
  asm volatile("s_wait_asynccnt 0x0" ::: "memory");
}

// ---- 16-lane-group xor butterflies: pure-VALU permlane16 when available ----
#if defined(__has_builtin)
#if __has_builtin(__builtin_amdgcn_permlane16)
#define HAVE_PERMLANE16 1
#endif
#endif

__device__ __forceinline__ float lanexor16(float v, int lo, int hi, int k) {
#ifdef HAVE_PERMLANE16
  int i = __builtin_bit_cast(int, v);
  i = __builtin_amdgcn_permlane16(i, i, lo, hi, false, false);
  (void)k;
  return __builtin_bit_cast(float, i);
#else
  (void)lo; (void)hi;
  return __shfl_xor(v, k, 32);
#endif
}
// selector nibble j = j^k (per 16-lane row)
__device__ __forceinline__ float redmax16(float v) {
  v = fmaxf(v, lanexor16(v, (int)0xFEDCBA98u, (int)0x76543210u, 8));
  v = fmaxf(v, lanexor16(v, (int)0x32107654u, (int)0xBA98FEDCu, 4));
  v = fmaxf(v, lanexor16(v, (int)0x54761032u, (int)0xDCFE98BAu, 2));
  v = fmaxf(v, lanexor16(v, (int)0x67452301u, (int)0xEFCDAB89u, 1));
  return v;
}
__device__ __forceinline__ float redsum16(float v) {
  v += lanexor16(v, (int)0xFEDCBA98u, (int)0x76543210u, 8);
  v += lanexor16(v, (int)0x32107654u, (int)0xBA98FEDCu, 4);
  v += lanexor16(v, (int)0x54761032u, (int)0xDCFE98BAu, 2);
  v += lanexor16(v, (int)0x67452301u, (int)0xEFCDAB89u, 1);
  return v;
}

// ---------------- GEMM: Y = act(Xmap @ Wb^T + bias) (+resid) ----------------
// X:  [M][Kd] bf16 (rows optionally window-remapped)
// Wb: [N][Kd] bf16 (pre-transposed weights) -> LDS rows are contiguous 16B runs
// Block = 256 threads = 8 waves; tile 64x64x32; double-buffered async pipeline.
template<bool GELU, bool OUTBF16>
__global__ __launch_bounds__(256)
void gemm_kernel(const unsigned short* __restrict__ X,
                 const unsigned short* __restrict__ Wb,
                 const float* __restrict__ bias, const float* __restrict__ resid,
                 void* __restrict__ Yv, int M, int N, int Kd, int winT, int winW)
{
  __shared__ __align__(16) unsigned short As[2][64][32];  // [buf][m][k]
  __shared__ __align__(16) unsigned short Bs[2][64][32];  // [buf][n][k]

  const int tid  = threadIdx.x;
  const int lane = tid & 31;
  const int wave = tid >> 5;
  const int wr = wave & 3;          // M quadrant (4 x 16 rows)
  const int wc = wave >> 2;         // N half (2 x 32 cols)
  const int m0 = blockIdx.x * 64;
  const int n0 = blockIdx.y * 64;

  const int lrow = tid >> 2;        // 0..63
  const int lcol = (tid & 3) * 8;   // 0,8,16,24  (16B chunks)

  int gm = m0 + lrow;
  const unsigned short* xrow = nullptr;
  if (gm < M) {
    int src = gm;
    if (winW > 0) { int s = gm / winW; src = s * winT + (winT - winW) + (gm - s * winW); }
    xrow = X + (size_t)src * Kd + lcol;
  }
  const unsigned short* wrow = Wb + (size_t)(n0 + lrow) * Kd + lcol;
  const unsigned ldsA[2] = { (unsigned)(uintptr_t)&As[0][lrow][lcol],
                             (unsigned)(uintptr_t)&As[1][lrow][lcol] };
  const unsigned ldsB[2] = { (unsigned)(uintptr_t)&Bs[0][lrow][lcol],
                             (unsigned)(uintptr_t)&Bs[1][lrow][lcol] };

  v8f acc0 = {0.f,0.f,0.f,0.f,0.f,0.f,0.f,0.f};
  v8f acc1 = acc0;

  const int rA  = wr * 16 + (lane & 15);
  const int kh  = (lane >> 4) * 8;    // A-frag K sub-offset (ISA 7.12.2)
  const int nB  = wc * 32 + (lane & 15);
  const int kh2 = (lane >> 4) * 16;   // unused; B-frags use kh as well

  // prologue: DMA tile 0
  if (xrow) async_b128(ldsA[0], xrow);
  async_b128(ldsB[0], wrow);
  wait_async0();
  __syncthreads();

  const int nk = Kd >> 5;
  for (int i = 0; i < nk; ++i) {
    const int cur = i & 1;
    const int nxt = cur ^ 1;
    if (i + 1 < nk) {                 // DMA tile i+1 while computing tile i
      const int kk = (i + 1) << 5;
      if (xrow) async_b128(ldsA[nxt], xrow + kk);
      async_b128(ldsB[nxt], wrow + kk);
    }
    ABfrag a, bb0, bb1;
    a.u4[0]   = *(const uint4*)&As[cur][rA][kh];
    a.u4[1]   = *(const uint4*)&As[cur][rA][16 + kh];
    bb0.u4[0] = *(const uint4*)&Bs[cur][nB][kh];
    bb0.u4[1] = *(const uint4*)&Bs[cur][nB][16 + kh];
    bb1.u4[0] = *(const uint4*)&Bs[cur][nB + 16][kh];
    bb1.u4[1] = *(const uint4*)&Bs[cur][nB + 16][16 + kh];
    acc0 = wmma_bf16(a, bb0, acc0);
    acc1 = wmma_bf16(a, bb1, acc1);
    wait_async0();                    // tile i+1 landed
    __syncthreads();
  }

  const int rowBase = m0 + wr * 16 + (lane >> 4) * 8;  // C-layout row mapping
  const int c0 = n0 + wc * 32 + (lane & 15);
  const float bv0 = bias ? bias[c0]      : 0.f;
  const float bv1 = bias ? bias[c0 + 16] : 0.f;
  float* Yf = (float*)Yv;
  unsigned short* Yh = (unsigned short*)Yv;
#pragma unroll
  for (int r = 0; r < 8; ++r) {
    int row = rowBase + r;
    if (row < M) {
      float y0 = acc0[r] + bv0;
      float y1 = acc1[r] + bv1;
      if (GELU) { y0 = gelu_exact(y0); y1 = gelu_exact(y1); }
      if (resid) {
        y0 += resid[(size_t)row * N + c0];
        y1 += resid[(size_t)row * N + c0 + 16];
      }
      if (OUTBF16) {
        Yh[(size_t)row * N + c0]      = f2bf(y0);
        Yh[(size_t)row * N + c0 + 16] = f2bf(y1);
      } else {
        Yf[(size_t)row * N + c0]      = y0;
        Yf[(size_t)row * N + c0 + 16] = y1;
      }
    }
  }
}

// ---------------- Flash attention: one wave per (seq, head, q-tile) -------
// Q/K/V: [nSeq, L, C] bf16 (produced by OUTBF16 GEMMs); O: f32. scale = 1/8.
__global__ __launch_bounds__(256)
void flash_kernel(const unsigned short* __restrict__ Qm,
                  const unsigned short* __restrict__ Km,
                  const unsigned short* __restrict__ Vm,
                  float* __restrict__ Om,
                  int nSeq, int L, float scale)
{
  __shared__ __align__(16) unsigned short Pt[8][16][32]; // per-wave P tile (bf16)

  const int lane = threadIdx.x & 31;
  const int wave = threadIdx.x >> 5;
  const int nQt  = (L + 15) >> 4;
  const int unit = blockIdx.x * 8 + wave;
  if (unit >= nSeq * NH * nQt) return;
  const int qt  = unit % nQt;
  const int sh  = unit / nQt;
  const int hh  = sh % NH;
  const int seq = sh / NH;
  const size_t base = (size_t)seq * L * C + (size_t)hh * HD;

  const int l15 = lane & 15;
  const int lg  = lane >> 4;

  // Q: 16x64 as two 16x32 A-frags; raw 16B loads, no conversion
  ABfrag qa[2];
  const int rQ = qt * 16 + l15;
#pragma unroll
  for (int f = 0; f < 2; ++f) {
    if (rQ < L) {
      const unsigned short* p = Qm + base + (size_t)rQ * C + f * 32 + lg * 8;
      qa[f].u4[0] = *(const uint4*)p;
      qa[f].u4[1] = *(const uint4*)(p + 16);
    } else {
      qa[f].u4[0] = make_uint4(0,0,0,0);
      qa[f].u4[1] = make_uint4(0,0,0,0);
    }
  }

  v8f o[4];
#pragma unroll
  for (int nf = 0; nf < 4; ++nf) o[nf] = (v8f){0.f,0.f,0.f,0.f,0.f,0.f,0.f,0.f};
  float mrow[8], lsum[8];
#pragma unroll
  for (int r = 0; r < 8; ++r) { mrow[r] = -1e30f; lsum[r] = 0.f; }

  const int nKt = (L + 31) >> 5;
  for (int kt = 0; kt < nKt; ++kt) {
    v8f s0 = (v8f){0.f,0.f,0.f,0.f,0.f,0.f,0.f,0.f};
    v8f s1 = s0;
    // scores: S = Q (16x64) x K^T (64x32)
#pragma unroll
    for (int cch = 0; cch < 2; ++cch) {
      int key = kt * 32 + cch * 16 + l15;
#pragma unroll
      for (int f = 0; f < 2; ++f) {
        ABfrag kb;
        if (key < L) {
          const unsigned short* p = Km + base + (size_t)key * C + f * 32 + lg * 8;
          kb.u4[0] = *(const uint4*)p;
          kb.u4[1] = *(const uint4*)(p + 16);
        } else {
          kb.u4[0] = make_uint4(0,0,0,0);
          kb.u4[1] = make_uint4(0,0,0,0);
        }
        if (cch == 0) s0 = wmma_bf16(qa[f], kb, s0);
        else          s1 = wmma_bf16(qa[f], kb, s1);
      }
    }
    // scale + column mask
    int c0 = kt * 32 + l15;
    int c1 = c0 + 16;
#pragma unroll
    for (int r = 0; r < 8; ++r) {
      s0[r] = (c0 < L) ? s0[r] * scale : -1e30f;
      s1[r] = (c1 < L) ? s1[r] * scale : -1e30f;
    }
    // online softmax (row r lives on register r across a 16-lane group)
#pragma unroll
    for (int r = 0; r < 8; ++r) {
      float mx   = redmax16(fmaxf(s0[r], s1[r]));
      float mnew = fmaxf(mrow[r], mx);
      float fac  = __expf(mrow[r] - mnew);
      float p0   = __expf(s0[r] - mnew);
      float p1   = __expf(s1[r] - mnew);
      float sm   = redsum16(p0 + p1);
      lsum[r] = lsum[r] * fac + sm;
      mrow[r] = mnew;
      s0[r] = p0; s1[r] = p1;
#pragma unroll
      for (int nf = 0; nf < 4; ++nf) o[nf][r] *= fac;
    }
    // P: C-layout -> LDS -> A-frag layout (same-wave DS ops stay ordered)
#pragma unroll
    for (int r = 0; r < 8; ++r) {
      Pt[wave][lg * 8 + r][l15]      = f2bf(s0[r]);
      Pt[wave][lg * 8 + r][l15 + 16] = f2bf(s1[r]);
    }
    ABfrag pa;
    pa.u4[0] = *(const uint4*)&Pt[wave][l15][lg * 16];
    pa.u4[1] = *(const uint4*)&Pt[wave][l15][lg * 16 + 8];
    // O += P (16x32) x V (32x64)
#pragma unroll
    for (int nf = 0; nf < 4; ++nf) {
      ABfrag vb;
      int d = nf * 16 + l15;
#pragma unroll
      for (int j = 0; j < 16; ++j) {
        int key = kt * 32 + lg * 16 + j;
        vb.s[j] = (key < L) ? Vm[base + (size_t)key * C + d] : (unsigned short)0;
      }
      o[nf] = wmma_bf16(pa, vb, o[nf]);
    }
  }

#pragma unroll
  for (int r = 0; r < 8; ++r) {
    int row = qt * 16 + lg * 8 + r;
    if (row < L) {
      float inv = (lsum[r] > 0.f) ? 1.f / lsum[r] : 0.f;
      float* op = Om + base + (size_t)row * C;
      op[0  + l15] = o[0][r] * inv;
      op[16 + l15] = o[1][r] * inv;
      op[32 + l15] = o[2][r] * inv;
      op[48 + l15] = o[3][r] * inv;
    }
  }
}

// ---------------- LayerNorm with gather mappings (bf16 output) ----------------
__device__ __forceinline__ float block_allreduce(float v, float* sm) {
  int lane = threadIdx.x & 31, wave = threadIdx.x >> 5;
#pragma unroll
  for (int off = 16; off > 0; off >>= 1) v += __shfl_xor(v, off, 32);
  __syncthreads();
  if (lane == 0) sm[wave] = v;
  __syncthreads();
  float t = (threadIdx.x < 8) ? sm[threadIdx.x] : 0.f;
#pragma unroll
  for (int off = 4; off > 0; off >>= 1) t += __shfl_xor(t, off, 32);
  if (threadIdx.x == 0) sm[0] = t;
  __syncthreads();
  return sm[0];
}

// mode 0: identity on src0; mode 1: temporal view of x[:,1:];
// mode 2: spatial view (cls from src0=x, patches from src1=xt2)
__global__ __launch_bounds__(256)
void ln_kernel(const float* __restrict__ src0, const float* __restrict__ src1,
               const float* __restrict__ gam, const float* __restrict__ bet,
               unsigned short* __restrict__ Y, int mode)
{
  __shared__ float sm[8];
  const int m = blockIdx.x;
  const int tid = threadIdx.x;
  const float* src;
  if (mode == 0) {
    src = src0 + (size_t)m * C;
  } else if (mode == 1) {
    int b = m / (KTOK * TT);
    int r = m - b * (KTOK * TT);
    src = src0 + ((size_t)b * (KTOK * TT + 1) + 1 + r) * C;
  } else {
    int bt = m / LSP;
    int i  = m - bt * LSP;
    int b  = bt >> 4;
    int t  = bt & 15;
    if (i == 0) src = src0 + (size_t)b * (KTOK * TT + 1) * C;
    else        src = src1 + ((size_t)b * (KTOK * TT) + (size_t)(i - 1) * TT + t) * C;
  }
  float x0 = src[tid], x1 = src[tid + 256], x2 = src[tid + 512];
  float mean = block_allreduce(x0 + x1 + x2, sm) * (1.f / C);
  float d0 = x0 - mean, d1 = x1 - mean, d2 = x2 - mean;
  float var = block_allreduce(d0*d0 + d1*d1 + d2*d2, sm) * (1.f / C);
  float inv = rsqrtf(var + 1e-5f);
  unsigned short* yp = Y + (size_t)m * C;
  yp[tid]       = f2bf(d0 * inv * gam[tid]       + bet[tid]);
  yp[tid + 256] = f2bf(d1 * inv * gam[tid + 256] + bet[tid + 256]);
  yp[tid + 512] = f2bf(d2 * inv * gam[tid + 512] + bet[tid + 512]);
}

// ---------------- conversion kernels ----------------
// weight transpose+convert: W [K][N] f32 (cnt matrices) -> Wb [N][K] bf16
__global__ void wcvtT_kernel(const float* __restrict__ W, unsigned short* __restrict__ Wb,
                             int K, int N, int cnt) {
  int i = blockIdx.x * 256 + threadIdx.x;
  if (i >= cnt * K * N) return;
  int per = K * N;
  int mat = i / per;
  int rem = i - mat * per;
  int n = rem / K;
  int k = rem - n * K;
  Wb[i] = f2bf(W[(size_t)mat * per + (size_t)k * N + n]);
}

__global__ void cvt_kernel(const float* __restrict__ in, unsigned short* __restrict__ outb, int n) {
  int i = blockIdx.x * 256 + threadIdx.x;
  if (i < n) outb[i] = f2bf(in[i]);
}

// ---------------- small elementwise kernels (f32 master path) ----------------
__global__ void mix_kernel(float* __restrict__ o, const float* __restrict__ p,
                           int nSeq, int Lbig, int off, int Wsm) {
  int idx = blockIdx.x * blockDim.x + threadIdx.x;
  if (idx >= nSeq * Wsm * C) return;
  int c = idx % C;
  int rem = idx / C;
  int t = rem % Wsm;
  int s = rem / Wsm;
  size_t oi = ((size_t)s * Lbig + off + t) * C + c;
  o[oi] = 0.5f * o[oi] + 0.5f * p[idx];
}

__global__ void addx_kernel(float* __restrict__ xt2, const float* __restrict__ x) {
  int idx = blockIdx.x * blockDim.x + threadIdx.x;
  if (idx >= M_T * C) return;
  int c = idx % C;
  int row = idx / C;
  int b = row / (KTOK * TT);
  int r = row - b * (KTOK * TT);
  xt2[idx] += x[((size_t)b * (KTOK * TT + 1) + 1 + r) * C + c];
}

__global__ void cls_mean_kernel(const float* __restrict__ osb, float* __restrict__ cls) {
  int idx = blockIdx.x * blockDim.x + threadIdx.x;
  if (idx >= BATCH * C) return;
  int b = idx / C, c = idx % C;
  float s = 0.f;
#pragma unroll
  for (int t = 0; t < TT; ++t)
    s += osb[((size_t)(b * TT + t) * LSP) * C + c];
  cls[idx] = s * (1.f / TT);
}

__global__ void build_xn_kernel(const float* __restrict__ x, const float* __restrict__ cls,
                                const float* __restrict__ xt2, const float* __restrict__ osb,
                                float* __restrict__ xn) {
  int idx = blockIdx.x * blockDim.x + threadIdx.x;
  if (idx >= M_F * C) return;
  int c = idx % C;
  int row = idx / C;
  int b = row / (KTOK * TT + 1);
  int j = row - b * (KTOK * TT + 1);
  float v;
  if (j == 0) {
    v = x[idx] + cls[b * C + c];
  } else {
    int k = (j - 1) / TT;
    int t = (j - 1) % TT;
    v = xt2[((size_t)b * (KTOK * TT) + (j - 1)) * C + c]
      + osb[(((size_t)(b * TT + t)) * LSP + 1 + k) * C + c];
  }
  xn[idx] = v;
}

// ---------------- launcher ----------------
extern "C" void kernel_launch(void* const* d_in, const int* in_sizes, int n_in,
                              void* d_out, int out_size, void* d_ws, size_t ws_size,
                              hipStream_t stream)
{
  (void)in_sizes; (void)n_in; (void)out_size; (void)ws_size;
  const float* x     = (const float*)d_in[0];
  const float* ln1_g = (const float*)d_in[1];
  const float* ln1_b = (const float*)d_in[2];
  const float* tln_g = (const float*)d_in[3];
  const float* tln_b = (const float*)d_in[4];
  const float* ln2_g = (const float*)d_in[5];
  const float* ln2_b = (const float*)d_in[6];
  const float* Ws    = (const float*)d_in[7];
  const float* bs    = (const float*)d_in[8];
  const float* Wt    = (const float*)d_in[9];
  const float* bt    = (const float*)d_in[10];
  const float* Wtp   = (const float*)d_in[11];
  const float* btp   = (const float*)d_in[12];
  const float* Wfc   = (const float*)d_in[13];
  const float* bfc   = (const float*)d_in[14];
  const float* W1    = (const float*)d_in[15];
  const float* b1    = (const float*)d_in[16];
  const float* W2    = (const float*)d_in[17];
  const float* b2    = (const float*)d_in[18];
  float* out = (float*)d_out;

  const size_t CC = (size_t)C * C;
  const size_t S_MS  = (size_t)M_S * C;
  const size_t S_MT  = (size_t)M_T * C;
  const size_t S_MF  = (size_t)M_F * C;
  const size_t S_M4  = (size_t)(NSEQT * 4) * C;
  const size_t S_M8  = (size_t)(NSEQT * 8) * C;
  const size_t S_MID = (size_t)3137 * 4 * C;

  // f32 master buffers
  float* wsf = (float*)d_ws;
  float* bufAO  = wsf; wsf += S_MS;               // attention output
  float* bufO4  = wsf; wsf += S_M4;
  float* bufO8  = wsf; wsf += S_M8;
  float* bufXT2 = wsf; wsf += S_MT;
  float* bufXN  = wsf; wsf += S_MF;
  float* bufOS  = wsf; wsf += S_MS;               // spatial proj output
  float* bufCLS = wsf; wsf += (size_t)BATCH * C;
  // bf16 buffers (16B-aligned: all sizes are multiples of C)
  unsigned short* wsb = (unsigned short*)wsf;
  unsigned short* bufHb  = wsb; wsb += S_MS;      // LN(temporal); reused: o16p bf16
  unsigned short* bufHSb = wsb; wsb += S_MS;      // LN(spatial); reused: h2
  unsigned short* bufQb  = wsb; wsb += S_MS;
  unsigned short* bufKb  = wsb; wsb += S_MS;
  unsigned short* bufVb  = wsb; wsb += S_MS;
  unsigned short* bufAb  = wsb; wsb += S_MS;      // bf16 copy of attention out
  unsigned short* bufMID = wsb; wsb += S_MID;     // MLP hidden (bf16)
  unsigned short* wpool  = wsb; wsb += 25 * CC;   // transposed bf16 weights

  unsigned short* WsT  = wpool + 0;        // 4 x CC
  unsigned short* WtT  = wpool + 4  * CC;  // 9 x CC
  unsigned short* WtpT = wpool + 13 * CC;  // 3 x CC
  unsigned short* WfcT = wpool + 16 * CC;  // 1 x CC
  unsigned short* W1T  = wpool + 17 * CC;  // C x 4C -> 4CC
  unsigned short* W2T  = wpool + 21 * CC;  // 4C x C -> 4CC

  auto gemm = [&](const unsigned short* X, const unsigned short* W, const float* b,
                  const float* res, void* Y, int M, int N, int Kd,
                  int winT, int winW, bool act, bool outbf) {
    dim3 grid((M + 63) / 64, N / 64);
    if (act) {
      if (outbf) gemm_kernel<true , true ><<<grid, 256, 0, stream>>>(X, W, b, res, Y, M, N, Kd, winT, winW);
      else       gemm_kernel<true , false><<<grid, 256, 0, stream>>>(X, W, b, res, Y, M, N, Kd, winT, winW);
    } else {
      if (outbf) gemm_kernel<false, true ><<<grid, 256, 0, stream>>>(X, W, b, res, Y, M, N, Kd, winT, winW);
      else       gemm_kernel<false, false><<<grid, 256, 0, stream>>>(X, W, b, res, Y, M, N, Kd, winT, winW);
    }
  };
  auto flash = [&](const unsigned short* Q, const unsigned short* Kq,
                   const unsigned short* V, float* O, int nSeq, int L) {
    int nQt = (L + 15) / 16;
    int units = nSeq * NH * nQt;
    flash_kernel<<<(units + 7) / 8, 256, 0, stream>>>(Q, Kq, V, O, nSeq, L, 0.125f);
  };
  auto cvt = [&](const float* in, unsigned short* ob, size_t n) {
    cvt_kernel<<<(int)((n + 255) / 256), 256, 0, stream>>>(in, ob, (int)n);
  };

  // 0) transpose+convert all weights to bf16 [N][K] once per call
  wcvtT_kernel<<<(int)((4*CC + 255)/256), 256, 0, stream>>>(Ws,  WsT,  C,   C,   4);
  wcvtT_kernel<<<(int)((9*CC + 255)/256), 256, 0, stream>>>(Wt,  WtT,  C,   C,   9);
  wcvtT_kernel<<<(int)((3*CC + 255)/256), 256, 0, stream>>>(Wtp, WtpT, C,   C,   3);
  wcvtT_kernel<<<(int)((1*CC + 255)/256), 256, 0, stream>>>(Wfc, WfcT, C,   C,   1);
  wcvtT_kernel<<<(int)((4*CC + 255)/256), 256, 0, stream>>>(W1,  W1T,  C,   4*C, 1);
  wcvtT_kernel<<<(int)((4*CC + 255)/256), 256, 0, stream>>>(W2,  W2T,  4*C, C,   1);

  // 1) temporal LN: h = LN(x[:,1:]) -> bf16
  ln_kernel<<<M_T, 256, 0, stream>>>(x, nullptr, tln_g, tln_b, bufHb, 1);

  // 2) temporal scale T=4 (last 4 tokens of each window)
  const int M4 = NSEQT * 4;
  gemm(bufHb, WtT + 0*CC, bt + 0*C, nullptr, bufQb, M4, C, C, TT, 4, false, true);
  gemm(bufHb, WtT + 1*CC, bt + 1*C, nullptr, bufKb, M4, C, C, TT, 4, false, true);
  gemm(bufHb, WtT + 2*CC, bt + 2*C, nullptr, bufVb, M4, C, C, TT, 4, false, true);
  flash(bufQb, bufKb, bufVb, bufAO, NSEQT, 4);
  cvt(bufAO, bufAb, (size_t)M4 * C);
  gemm(bufAb, WtpT + 0*CC, btp + 0*C, nullptr, bufO4, M4, C, C, 0, 0, false, false);

  // 3) temporal scale T=8 (last 8 tokens)
  const int M8 = NSEQT * 8;
  gemm(bufHb, WtT + 3*CC, bt + 3*C, nullptr, bufQb, M8, C, C, TT, 8, false, true);
  gemm(bufHb, WtT + 4*CC, bt + 4*C, nullptr, bufKb, M8, C, C, TT, 8, false, true);
  gemm(bufHb, WtT + 5*CC, bt + 5*C, nullptr, bufVb, M8, C, C, TT, 8, false, true);
  flash(bufQb, bufKb, bufVb, bufAO, NSEQT, 8);
  mix_kernel<<<(NSEQT*4*C + 255) / 256, 256, 0, stream>>>(bufAO, bufO4, NSEQT, 8, 4, 4);
  cvt(bufAO, bufAb, (size_t)M8 * C);
  gemm(bufAb, WtpT + 1*CC, btp + 1*C, nullptr, bufO8, M8, C, C, 0, 0, false, false);

  // 4) temporal scale T=16 (full windows)
  gemm(bufHb, WtT + 6*CC, bt + 6*C, nullptr, bufQb, M_T, C, C, 0, 0, false, true);
  gemm(bufHb, WtT + 7*CC, bt + 7*C, nullptr, bufKb, M_T, C, C, 0, 0, false, true);
  gemm(bufHb, WtT + 8*CC, bt + 8*C, nullptr, bufVb, M_T, C, C, 0, 0, false, true);
  flash(bufQb, bufKb, bufVb, bufAO, NSEQT, 16);
  mix_kernel<<<(NSEQT*8*C + 255) / 256, 256, 0, stream>>>(bufAO, bufO8, NSEQT, 16, 8, 8);
  cvt(bufAO, bufAb, (size_t)M_T * C);
  gemm(bufAb, WtpT + 2*CC, btp + 2*C, nullptr, bufHb /*o16p bf16*/, M_T, C, C, 0, 0, false, true);
  gemm(bufHb, WfcT, bfc, nullptr, bufXT2, M_T, C, C, 0, 0, false, false);
  addx_kernel<<<(M_T*C + 255) / 256, 256, 0, stream>>>(bufXT2, x);

  // 5) spatial attention over 197 tokens
  ln_kernel<<<M_S, 256, 0, stream>>>(x, bufXT2, ln1_g, ln1_b, bufHSb, 2);
  gemm(bufHSb, WsT + 0*CC, bs + 0*C, nullptr, bufQb, M_S, C, C, 0, 0, false, true);
  gemm(bufHSb, WsT + 1*CC, bs + 1*C, nullptr, bufKb, M_S, C, C, 0, 0, false, true);
  gemm(bufHSb, WsT + 2*CC, bs + 2*C, nullptr, bufVb, M_S, C, C, 0, 0, false, true);
  flash(bufQb, bufKb, bufVb, bufAO, NSEQS, LSP);
  cvt(bufAO, bufAb, (size_t)M_S * C);
  gemm(bufAb, WsT + 3*CC, bs + 3*C, nullptr, bufOS, M_S, C, C, 0, 0, false, false);
  cls_mean_kernel<<<(BATCH*C + 255) / 256, 256, 0, stream>>>(bufOS, bufCLS);
  build_xn_kernel<<<(M_F*C + 255) / 256, 256, 0, stream>>>(x, bufCLS, bufXT2, bufOS, bufXN);

  // 6) MLP (per-batch chunks), resid fused into final GEMM
  ln_kernel<<<M_F, 256, 0, stream>>>(bufXN, nullptr, ln2_g, ln2_b, bufHSb, 0);
  for (int b = 0; b < BATCH; ++b) {
    size_t off = (size_t)b * 3137 * C;
    gemm(bufHSb + off, W1T, b1, nullptr, bufMID, 3137, 4*C, C, 0, 0, true, true);
    gemm(bufMID, W2T, b2, bufXN + off, out + off, 3137, C, 4*C, 0, 0, false, false);
  }
}
